// RnnDecoder_7473243095645
// MI455X (gfx1250) — compile-verified, matrix operand order checked by reading, not run on hardware
//
#include <hip/hip_runtime.h>
#include <hip/hip_bf16.h>

typedef __bf16 bf16_t;
typedef __attribute__((ext_vector_type(16))) __bf16 v16bf;
typedef __attribute__((ext_vector_type(8)))  __bf16 v8bf;
typedef __attribute__((ext_vector_type(8)))  float  v8f;
typedef __attribute__((ext_vector_type(4)))  float  vf4;

// Problem dimensions (fixed by the reference)
#define LSEQ 512
#define BAT  64
#define ADIM 256
#define IDIM 256
#define HID  512
#define GATES 2048              // 4*HID
#define WLD  512                // W_ih row stride = I + A ; W_hh row stride = H

// Recurrence kernel geometry
#define RC_BLOCKS 32            // one per hidden 16-col tile
#define RC_THREADS 128          // 4 waves = 4 batch tiles
#define NFRAG 24                // 16 K-frags of W_hh (K=512) + 8 K-frags of W_a (K=256)
#define RC_LDS_BYTES (4 * NFRAG * 32 * (int)sizeof(v16bf))   // 96KB

__device__ __forceinline__ float sigm_f(float x) {
    return 1.0f / (1.0f + __expf(-x));
}
__device__ __forceinline__ float tanh_f(float x) {
    float e = __expf(-2.0f * x);
    return (1.0f - e) / (1.0f + e);
}

// Build a 16x32 bf16 A-fragment per ISA layout from a row-major fp32 matrix.
// lane L: row = row0 + (L&15); elems 0..7 at k = k0 + 8*(L>>4) + j ; 8..15 at +16.
__device__ __forceinline__ v16bf load_frag_f32(const float* __restrict__ src, int ld,
                                               int row0, int k0, int lane) {
    const float* p = src + (size_t)(row0 + (lane & 15)) * ld + k0 + 8 * (lane >> 4);
    v16bf v;
#pragma unroll
    for (int j = 0; j < 8; ++j) v[j] = (__bf16)p[j];
#pragma unroll
    for (int j = 0; j < 8; ++j) v[8 + j] = (__bf16)p[16 + j];
    return v;
}

// Same fragment layout from a row-major bf16 matrix (two b128 loads, no cvt).
__device__ __forceinline__ v16bf load_frag_bf16(const bf16_t* __restrict__ src, int ld,
                                                int row0, int k0, int lane) {
    const bf16_t* p = src + (size_t)(row0 + (lane & 15)) * ld + k0 + 8 * (lane >> 4);
    v8bf lo = *(const v8bf*)(p);
    v8bf hi = *(const v8bf*)(p + 16);
    return __builtin_shufflevector(lo, hi, 0, 1, 2, 3, 4, 5, 6, 7,
                                   8, 9, 10, 11, 12, 13, 14, 15);
}

// ---------------------------------------------------------------------------
// Kernel 1a: init barrier state + convert h0 -> bf16 ping buffer
// ---------------------------------------------------------------------------
__global__ void lstm_init_kernel(unsigned* __restrict__ bar,
                                 const float* __restrict__ h0,
                                 bf16_t* __restrict__ hb0) {
    int idx = blockIdx.x * blockDim.x + threadIdx.x;
    if (idx < 2) bar[idx] = 0u;
    for (int i = idx; i < BAT * HID; i += gridDim.x * blockDim.x)
        hb0[i] = (__bf16)h0[i];
}

// ---------------------------------------------------------------------------
// Kernel 1b: convert action_features (L,B,A) fp32 -> bf16 once (so the serial
// scan never touches fp32 act data and does zero act conversions).
// 4 floats per thread: one b128 load -> two v_cvt_pk_bf16_f32 -> one b64 store.
// ---------------------------------------------------------------------------
__global__ void lstm_cvt_act_kernel(const float* __restrict__ act,
                                    bf16_t* __restrict__ act_bf) {
    const int n4 = (LSEQ * BAT * ADIM) / 4;
    for (int i = blockIdx.x * blockDim.x + threadIdx.x; i < n4;
         i += gridDim.x * blockDim.x) {
        vf4 v = *(const vf4*)(act + (size_t)i * 4);
        bf16_t* o = act_bf + (size_t)i * 4;
        o[0] = (__bf16)v.x; o[1] = (__bf16)v.y;
        o[2] = (__bf16)v.z; o[3] = (__bf16)v.w;
    }
}

// ---------------------------------------------------------------------------
// Kernel 2: static_gates[b,g] = input_feature @ W_i.T + b_ih + b_hh   (64x2048)
// One 16x16 tile per wave, K=256 via 8 bf16 WMMAs.
// ---------------------------------------------------------------------------
__global__ void __launch_bounds__(256)
lstm_static_kernel(const float* __restrict__ inp, const float* __restrict__ W_ih,
                   const float* __restrict__ b_ih, const float* __restrict__ b_hh,
                   float* __restrict__ statg) {
    const int tid  = threadIdx.x;
    const int lane = tid & 31;
    const int wave = tid >> 5;
    const int tile = blockIdx.x * 8 + wave;   // 0..511
    const int m = tile & 3;                   // batch tile (B=64 -> 4)
    const int n = tile >> 2;                  // gate tile  (G=2048 -> 128)
    const int lhalf = lane >> 4;
    const int l15   = lane & 15;

    v8f acc = {};
#pragma unroll
    for (int kf = 0; kf < 8; ++kf) {
        v16bf a = load_frag_f32(inp,  IDIM, m * 16, kf * 32, lane);
        v16bf b = load_frag_f32(W_ih, WLD,  n * 16, kf * 32, lane);  // I-part cols 0..255
        acc = __builtin_amdgcn_wmma_f32_16x16x32_bf16(false, a, false, b,
                                                      (short)0, acc, false, false);
    }
    const int gcol = n * 16 + l15;
    const float bias = b_ih[gcol] + b_hh[gcol];
#pragma unroll
    for (int j = 0; j < 8; ++j) {
        int b = m * 16 + j + 8 * lhalf;
        statg[(size_t)b * GATES + gcol] = acc[j] + bias;
    }
}

// ---------------------------------------------------------------------------
// Kernel 3: persistent LSTM scan. 32 WGs x 128 threads; WG = hidden tile n,
// wave = batch tile m. W_hh|W_a gate-slices pre-swizzled into 96KB LDS as
// ready-to-use B fragments. c stays in VGPRs; h ping-pongs via bf16 L2 bufs.
// Grid sync = sense-reversal barrier in workspace.
// ---------------------------------------------------------------------------
__global__ void __launch_bounds__(RC_THREADS)
lstm_recur_kernel(const bf16_t* __restrict__ act_bf, // (L,B,A) bf16 in ws
                  const float* __restrict__ c0,      // (B,H)
                  const float* __restrict__ W_hh,    // (4H,H)
                  const float* __restrict__ W_ih,    // (4H,I+A)
                  const float* __restrict__ statg,   // (B,4H) fp32 in ws
                  bf16_t* __restrict__ hbuf0, bf16_t* __restrict__ hbuf1,
                  unsigned* __restrict__ bar,
                  float* __restrict__ out) {
    extern __shared__ v16bf lds[];   // [gate][frag][lane] : 4*24*32 entries
    const int tid   = threadIdx.x;
    const int lane  = tid & 31;
    const int wave  = tid >> 5;      // batch tile 0..3
    const int nblk  = blockIdx.x;    // hidden tile 0..31
    const int lhalf = lane >> 4;
    const int l15   = lane & 15;

    // ---- Preload & swizzle B fragments: W_hh (frags 0..15), W_a (frags 16..23)
    for (int e = tid; e < 4 * NFRAG * 32; e += RC_THREADS) {
        const int elane = e & 31;
        const int kf = (e >> 5) % NFRAG;
        const int q  = (e >> 5) / NFRAG;                     // gate 0..3
        const int g  = q * HID + nblk * 16 + (elane & 15);   // W row (gate index)
        const float* src;
        int col;
        if (kf < 16) { src = W_hh + (size_t)g * HID; col = kf * 32 + 8 * (elane >> 4); }
        else         { src = W_ih + (size_t)g * WLD; col = IDIM + (kf - 16) * 32 + 8 * (elane >> 4); }
        v16bf v;
#pragma unroll
        for (int j = 0; j < 8; ++j) v[j] = (__bf16)src[col + j];
#pragma unroll
        for (int j = 0; j < 8; ++j) v[8 + j] = (__bf16)src[col + 16 + j];
        lds[e] = v;
    }
    __syncthreads();

    const int hcol = nblk * 16 + l15;   // hidden column this lane owns in C/D
    const int arow = wave * 16 + l15;   // batch row this lane loads for A frags

    // ---- Persistent state: c in regs, static gates in regs
    float c[8], sg[4][8], hl[8];
#pragma unroll
    for (int j = 0; j < 8; ++j) {
        const int b = wave * 16 + j + 8 * lhalf;
        c[j] = c0[(size_t)b * HID + hcol];
#pragma unroll
        for (int q = 0; q < 4; ++q)
            sg[q][j] = statg[(size_t)b * GATES + q * HID + hcol];
    }

    bf16_t* hb[2] = { hbuf0, hbuf1 };

    for (int l = 0; l < LSEQ; ++l) {
        const bf16_t* hp = hb[l & 1];
        bf16_t*       hn = hb[(l + 1) & 1];

        v8f acc[4];
#pragma unroll
        for (int q = 0; q < 4; ++q) {
            v8f t;
#pragma unroll
            for (int j = 0; j < 8; ++j) t[j] = sg[q][j];
            acc[q] = t;
        }

        // h @ W_hh.T : K = 512, A-frag reused across the 4 gates
#pragma unroll
        for (int kf = 0; kf < 16; ++kf) {
            v16bf a = load_frag_bf16(hp, HID, wave * 16, kf * 32, lane);
#pragma unroll
            for (int q = 0; q < 4; ++q)
                acc[q] = __builtin_amdgcn_wmma_f32_16x16x32_bf16(
                    false, a, false, lds[(q * NFRAG + kf) * 32 + lane],
                    (short)0, acc[q], false, false);
        }
        // act[l] @ W_a.T : K = 256 (fused act_gates), bf16 source, no cvt
        const bf16_t* al = act_bf + (size_t)l * BAT * ADIM;
#pragma unroll
        for (int kf = 0; kf < 8; ++kf) {
            v16bf a = load_frag_bf16(al, ADIM, wave * 16, kf * 32, lane);
#pragma unroll
            for (int q = 0; q < 4; ++q)
                acc[q] = __builtin_amdgcn_wmma_f32_16x16x32_bf16(
                    false, a, false, lds[(q * NFRAG + 16 + kf) * 32 + lane],
                    (short)0, acc[q], false, false);
        }

        if (l + 1 < LSEQ)   // pull next step's action rows toward the WGP
            __builtin_prefetch(act_bf + (size_t)(l + 1) * BAT * ADIM + (size_t)arow * ADIM, 0, 1);

        // LSTM cell elementwise + emit hs[l] (fp32) and next-h (bf16)
        float* hsout = out + (size_t)l * BAT * HID;
#pragma unroll
        for (int j = 0; j < 8; ++j) {
            const int b = wave * 16 + j + 8 * lhalf;
            const float ig = sigm_f(acc[0][j]);
            const float fg = sigm_f(acc[1][j]);
            const float gg = tanh_f(acc[2][j]);
            const float og = sigm_f(acc[3][j]);
            c[j] = fg * c[j] + ig * gg;
            const float h = og * tanh_f(c[j]);
            hl[j] = h;
            hsout[(size_t)b * HID + hcol] = h;
            hn[(size_t)b * HID + hcol] = (__bf16)h;
        }

        // ---- grid-wide sense-reversal barrier (32 WGs)
        __syncthreads();
        if (tid == 0) {
            __threadfence();
            unsigned gen  = __hip_atomic_load(&bar[1], __ATOMIC_ACQUIRE, __HIP_MEMORY_SCOPE_AGENT);
            unsigned prev = __hip_atomic_fetch_add(&bar[0], 1u, __ATOMIC_ACQ_REL, __HIP_MEMORY_SCOPE_AGENT);
            if (prev == (unsigned)(RC_BLOCKS - 1)) {
                __hip_atomic_store(&bar[0], 0u, __ATOMIC_RELAXED, __HIP_MEMORY_SCOPE_AGENT);
                __hip_atomic_fetch_add(&bar[1], 1u, __ATOMIC_RELEASE, __HIP_MEMORY_SCOPE_AGENT);
            } else {
                while (__hip_atomic_load(&bar[1], __ATOMIC_ACQUIRE, __HIP_MEMORY_SCOPE_AGENT) == gen)
                    __builtin_amdgcn_s_sleep(1);
            }
        }
        __syncthreads();
    }

    // ---- final h_n, c_n
    float* hn_out = out + (size_t)LSEQ * BAT * HID;
    float* cn_out = hn_out + BAT * HID;
#pragma unroll
    for (int j = 0; j < 8; ++j) {
        const int b = wave * 16 + j + 8 * lhalf;
        hn_out[(size_t)b * HID + hcol] = hl[j];
        cn_out[(size_t)b * HID + hcol] = c[j];
    }
}

// ---------------------------------------------------------------------------
extern "C" void kernel_launch(void* const* d_in, const int* in_sizes, int n_in,
                              void* d_out, int out_size, void* d_ws, size_t ws_size,
                              hipStream_t stream) {
    const float* act  = (const float*)d_in[0];  // (512,64,256)
    const float* inp  = (const float*)d_in[1];  // (64,256)
    const float* h0   = (const float*)d_in[2];  // (64,512)
    const float* c0   = (const float*)d_in[3];  // (64,512)
    const float* W_ih = (const float*)d_in[4];  // (2048,512)
    const float* W_hh = (const float*)d_in[5];  // (2048,512)
    const float* b_ih = (const float*)d_in[6];  // (2048)
    const float* b_hh = (const float*)d_in[7];  // (2048)
    float* out = (float*)d_out;

    char* ws = (char*)d_ws;
    unsigned* bar   = (unsigned*)ws;                                    // 256 B
    float*    stg   = (float*)(ws + 256);                               // 512 KB
    bf16_t*   hb0   = (bf16_t*)(ws + 256 + (size_t)BAT * GATES * 4);    // 64 KB
    bf16_t*   hb1   = hb0 + (size_t)BAT * HID;                          // 64 KB
    bf16_t*   actbf = hb1 + (size_t)BAT * HID;                          // 16.8 MB

    lstm_init_kernel<<<32, 256, 0, stream>>>(bar, h0, hb0);
    lstm_cvt_act_kernel<<<2048, 256, 0, stream>>>(act, actbf);
    lstm_static_kernel<<<64, 256, 0, stream>>>(inp, W_ih, b_ih, b_hh, stg);
    lstm_recur_kernel<<<RC_BLOCKS, RC_THREADS, RC_LDS_BYTES, stream>>>(
        actbf, c0, W_hh, W_ih, stg, hb0, hb1, bar, out);
}
